// EdgeGNNLayer_19086834664179
// MI455X (gfx1250) — matrix operand order repaired and checked by reading, not compiled, not run on hardware
//
#include <hip/hip_runtime.h>
#include <hip/hip_bf16.h>

#define HID 128
#define ATT 64

typedef __attribute__((ext_vector_type(16))) __bf16 v16bf;
typedef __attribute__((ext_vector_type(8)))  __bf16 v8bf;
typedef __attribute__((ext_vector_type(8)))  float  v8f;
typedef __attribute__((ext_vector_type(4)))  float  v4f;

// Compiler-only memory barrier: stops LICM from hoisting the (loop-invariant)
// LDS weight reads out of the persistent tile loop, which previously made the
// register allocator keep ~80 B-fragments live and spill them to scratch.
#define KEEP_LOADS_IN_LOOP() asm volatile("" ::: "memory")

// ---------------------------------------------------------------------------
// WMMA helper: D = A(16x32 bf16) * B(32x16 bf16) + C(16x16 f32)
// ---------------------------------------------------------------------------
static __device__ inline v8f wmma_bf16(v16bf a, v16bf b, v8f c) {
    return __builtin_amdgcn_wmma_f32_16x16x32_bf16(
        /*neg_a=*/false, a, /*neg_b=*/false, b,
        /*c_mod=*/(short)0, c, /*reuse_a=*/false, /*reuse_b=*/false);
}

// A-fragment (16x32 bf16, MxK) gathered from a row-major bf16 row.
// ISA layout: lanes 0-15 (half=0): K = {kc*32+0..7, kc*32+16..23}
//             lanes 16-31 (half=1): K = {kc*32+8..15, kc*32+24..31}
static __device__ inline v16bf load_afrag_bf(const __bf16* __restrict__ row,
                                             int kc, int half) {
    const v8bf lo = *(const v8bf*)(row + kc * 32 + half * 8);
    const v8bf hi = *(const v8bf*)(row + kc * 32 + 16 + half * 8);
    v16bf a;
#pragma unroll
    for (int i = 0; i < 8; ++i) { a[i] = lo[i]; a[i + 8] = hi[i]; }
    return a;
}

// B-fragment (32x16 bf16, KxN) from W stored row-major [N][K=HID] in LDS.
// Column n of B == row n of W (since B = W^T). Lane layout:
//   lanes 0-15:  N = n0+lane,    K = {kc*32+0..7, kc*32+16..23}
//   lanes 16-31: N = n0+lane-16, K = {kc*32+8..15, kc*32+24..31}
static __device__ inline v16bf load_bfrag_s(const __bf16* W, int n0, int kc,
                                            int lm, int half) {
    return load_afrag_bf(W + (n0 + lm) * HID, kc, half);
}

// Cooperative LDS staging: copy n_bf16 bf16 elements (multiple of 8) as uint4.
static __device__ inline void stage_lds(const __bf16* __restrict__ g, __bf16* s,
                                        int n_bf16) {
    const uint4* gs = (const uint4*)g;
    uint4*       ss = (uint4*)s;
    const int    n  = n_bf16 >> 3;
    for (int i = threadIdx.x; i < n; i += blockDim.x) ss[i] = gs[i];
}

// ---------------------------------------------------------------------------
// Prep kernels
// ---------------------------------------------------------------------------
__global__ void k_f32_to_bf16(const float* __restrict__ in,
                              __bf16* __restrict__ out, int n) {
    int i = blockIdx.x * blockDim.x + threadIdx.x;
    const int stride = gridDim.x * blockDim.x;
    for (; i < n; i += stride) out[i] = (__bf16)in[i];
}

__global__ void k_zero_f32(float* __restrict__ p, int n) {
    int i = blockIdx.x * blockDim.x + threadIdx.x;
    const int stride = gridDim.x * blockDim.x;
    for (; i < n; i += stride) p[i] = 0.0f;
}

// ---------------------------------------------------------------------------
// Line-graph kernel: attention gate + messages + scatter-add.
// Persistent wave32 per 16-line tile; 8 waves (256 threads) per block.
// All weight matrices staged once per block into LDS (80 KB of 320 KB/WGP).
// ---------------------------------------------------------------------------
__global__ __launch_bounds__(256, 1)
void k_line(const __bf16* __restrict__ eh, const __bf16* __restrict__ qr,
            const __bf16* __restrict__ Wsrc, const __bf16* __restrict__ Wdst,
            const __bf16* __restrict__ Wqr, const float* __restrict__ b_qr,
            const float* __restrict__ w_alpha, const float* __restrict__ b_alpha,
            const __bf16* __restrict__ Wmsg,
            const int* __restrict__ lsrc, const int* __restrict__ ldst,
            float* __restrict__ agg, int n_line) {
    __shared__ __align__(32) __bf16 sWsrc[ATT * HID];
    __shared__ __align__(32) __bf16 sWdst[ATT * HID];
    __shared__ __align__(32) __bf16 sWqr [ATT * HID];
    __shared__ __align__(32) __bf16 sWmsg[HID * HID];

    stage_lds(Wsrc, sWsrc, ATT * HID);
    stage_lds(Wdst, sWdst, ATT * HID);
    stage_lds(Wqr,  sWqr,  ATT * HID);
    stage_lds(Wmsg, sWmsg, HID * HID);
    __syncthreads();

    const int lane = threadIdx.x & 31;
    const int wave = threadIdx.x >> 5;
    const int half = lane >> 4;   // 0: K-low half of A frag; 1: K-high half
    const int lm   = lane & 15;
    const int wstride = gridDim.x * 8;
    const float balpha = b_alpha[0];

    for (int tile = blockIdx.x * 8 + wave; tile * 16 < n_line; tile += wstride) {
        KEEP_LOADS_IN_LOOP();            // re-read LDS weights this iteration
        const int base = tile * 16;      // n_line is a multiple of 16

        // Row indices: lane lm owns line (base+lm) for the A layout.
        const int src_i = lsrc[base + lm];
        const int dst_i = ldst[base + lm];

        // Gather A fragments for h_src, h_dst, h_qr (K = 128 -> 4 chunks).
        v16bf aS[4], aD[4], aQ[4];
        {
            const __bf16* rS = eh + (size_t)src_i * HID;
            const __bf16* rD = eh + (size_t)dst_i * HID;
            const __bf16* rQ = qr + (size_t)dst_i * HID;
#pragma unroll
            for (int kc = 0; kc < 4; ++kc) {
                aS[kc] = load_afrag_bf(rS, kc, half);
                aD[kc] = load_afrag_bf(rD, kc, half);
                aQ[kc] = load_afrag_bf(rQ, kc, half);
            }
        }

        // pre = hS@Wsrc^T + hD@Wdst^T + hQ@Wqr^T + b_qr  (16 x 64), folded
        // immediately into dot = sum_n relu(pre[m,n]) * w_alpha[n].
        v8f dotacc = {};
#pragma unroll
        for (int nt = 0; nt < 4; ++nt) {
            v8f c = {};
#pragma unroll
            for (int kc = 0; kc < 4; ++kc) {
                c = wmma_bf16(aS[kc], load_bfrag_s(sWsrc, nt * 16, kc, lm, half), c);
                c = wmma_bf16(aD[kc], load_bfrag_s(sWdst, nt * 16, kc, lm, half), c);
                c = wmma_bf16(aQ[kc], load_bfrag_s(sWqr,  nt * 16, kc, lm, half), c);
            }
            // C layout: vgpr j, lane l -> row m = j + 8*half, col n = nt*16+lm
            const int   n  = nt * 16 + lm;
            const float bq = b_qr[n];
            const float wa = w_alpha[n];
#pragma unroll
            for (int j = 0; j < 8; ++j) {
                float v = c[j] + bq;
                v = v > 0.0f ? v : 0.0f;
                dotacc[j] += v * wa;
            }
        }
        // Reduce over the 16 lanes of each half (xor masks 1..8 stay in-half).
#pragma unroll
        for (int off = 1; off < 16; off <<= 1) {
#pragma unroll
            for (int j = 0; j < 8; ++j)
                dotacc[j] += __shfl_xor(dotacc[j], off, 32);
        }
        // alpha for row m = j + 8*half  -- exactly the msg C-fragment layout.
        float alphav[8];
#pragma unroll
        for (int j = 0; j < 8; ++j)
            alphav[j] = 1.0f / (1.0f + __expf(-(dotacc[j] + balpha)));

        // dst index per fragment row (lane m holds row m's dst).
        int drow[8];
#pragma unroll
        for (int j = 0; j < 8; ++j)
            drow[j] = __shfl(dst_i, j + 8 * half, 32);

        KEEP_LOADS_IN_LOOP();            // don't pre-load Wmsg frags early

        // msg = alpha * (h_src @ Wmsg^T)  (16 x 128) + scatter-add into agg.
#pragma unroll
        for (int nt = 0; nt < 8; ++nt) {
            v8f m = {};
#pragma unroll
            for (int kc = 0; kc < 4; ++kc)
                m = wmma_bf16(aS[kc], load_bfrag_s(sWmsg, nt * 16, kc, lm, half), m);
            const int n = nt * 16 + lm;
#pragma unroll
            for (int j = 0; j < 8; ++j)
                atomicAdd(&agg[(size_t)drow[j] * HID + n], m[j] * alphav[j]);
        }
    }
}

// ---------------------------------------------------------------------------
// Output GEMM: out = agg @ W_out^T   ([n_edge,128] x [128,128])
// ---------------------------------------------------------------------------
__global__ __launch_bounds__(256, 1)
void k_out(const float* __restrict__ agg, const __bf16* __restrict__ Wout,
           float* __restrict__ out, int n_edge) {
    __shared__ __align__(32) __bf16 sWout[HID * HID];
    stage_lds(Wout, sWout, HID * HID);
    __syncthreads();

    const int lane = threadIdx.x & 31;
    const int wave = threadIdx.x >> 5;
    const int half = lane >> 4;
    const int lm   = lane & 15;
    const int wstride = gridDim.x * 8;

    for (int tile = blockIdx.x * 8 + wave; tile * 16 < n_edge; tile += wstride) {
        KEEP_LOADS_IN_LOOP();            // re-read LDS weights this iteration
        const int base = tile * 16;      // n_edge is a multiple of 16

        // Build bf16 A fragments from f32 agg rows (lane lm owns row base+lm).
        const float* row = agg + (size_t)(base + lm) * HID;
        v16bf a[4];
#pragma unroll
        for (int kc = 0; kc < 4; ++kc) {
            const v4f x0 = *(const v4f*)(row + kc * 32 + half * 8);
            const v4f x1 = *(const v4f*)(row + kc * 32 + half * 8 + 4);
            const v4f x2 = *(const v4f*)(row + kc * 32 + 16 + half * 8);
            const v4f x3 = *(const v4f*)(row + kc * 32 + 16 + half * 8 + 4);
            v16bf t;
#pragma unroll
            for (int i = 0; i < 4; ++i) {
                t[i]      = (__bf16)x0[i];
                t[i + 4]  = (__bf16)x1[i];
                t[i + 8]  = (__bf16)x2[i];
                t[i + 12] = (__bf16)x3[i];
            }
            a[kc] = t;
        }
#pragma unroll
        for (int nt = 0; nt < 8; ++nt) {
            v8f c = {};
#pragma unroll
            for (int kc = 0; kc < 4; ++kc)
                c = wmma_bf16(a[kc], load_bfrag_s(sWout, nt * 16, kc, lm, half), c);
            const int n = nt * 16 + lm;
#pragma unroll
            for (int j = 0; j < 8; ++j)
                out[(size_t)(base + j + 8 * half) * HID + n] = c[j];
        }
    }
}

// ---------------------------------------------------------------------------
// Host launcher
// ---------------------------------------------------------------------------
extern "C" void kernel_launch(void* const* d_in, const int* in_sizes, int n_in,
                              void* d_out, int out_size, void* d_ws, size_t ws_size,
                              hipStream_t stream) {
    const float* edge_hidden = (const float*)d_in[0];
    const float* q_rel_emb   = (const float*)d_in[1];
    const float* W_src       = (const float*)d_in[2];
    const float* W_dst       = (const float*)d_in[3];
    const float* W_qr        = (const float*)d_in[4];
    const float* b_qr        = (const float*)d_in[5];
    const float* w_alpha     = (const float*)d_in[6];
    const float* b_alpha     = (const float*)d_in[7];
    const float* W_msg       = (const float*)d_in[8];
    const float* W_out       = (const float*)d_in[9];
    const int*   line_src    = (const int*)d_in[10];
    const int*   line_dst    = (const int*)d_in[11];

    const int n_edge = in_sizes[0] / HID;
    const int n_line = in_sizes[10];
    float* out = (float*)d_out;

    // Workspace carve-out (256B aligned slices).
    char*  ws  = (char*)d_ws;
    size_t off = 0;
    auto carve = [&](size_t bytes) -> char* {
        char* p = ws + off;
        off += (bytes + 255) & ~(size_t)255;
        return p;
    };
    float*  agg     = (float*) carve((size_t)n_edge * HID * sizeof(float));
    __bf16* eh_bf   = (__bf16*)carve((size_t)n_edge * HID * sizeof(__bf16));
    __bf16* qr_bf   = (__bf16*)carve((size_t)n_edge * HID * sizeof(__bf16));
    __bf16* Wsrc_bf = (__bf16*)carve((size_t)ATT * HID * sizeof(__bf16));
    __bf16* Wdst_bf = (__bf16*)carve((size_t)ATT * HID * sizeof(__bf16));
    __bf16* Wqr_bf  = (__bf16*)carve((size_t)ATT * HID * sizeof(__bf16));
    __bf16* Wmsg_bf = (__bf16*)carve((size_t)HID * HID * sizeof(__bf16));
    __bf16* Wout_bf = (__bf16*)carve((size_t)HID * HID * sizeof(__bf16));

    const int threads = 256;
    auto blocks_for = [&](long long n) -> int {
        long long b = (n + threads - 1) / threads;
        return (int)(b < 4096 ? b : 4096);
    };

    // 1) zero accumulator
    k_zero_f32<<<blocks_for((long long)n_edge * HID), threads, 0, stream>>>(
        agg, n_edge * HID);

    // 2) bf16 conversions (weights + feature tables)
    k_f32_to_bf16<<<blocks_for((long long)n_edge * HID), threads, 0, stream>>>(
        edge_hidden, eh_bf, n_edge * HID);
    k_f32_to_bf16<<<blocks_for((long long)n_edge * HID), threads, 0, stream>>>(
        q_rel_emb, qr_bf, n_edge * HID);
    k_f32_to_bf16<<<blocks_for(ATT * HID), threads, 0, stream>>>(W_src, Wsrc_bf, ATT * HID);
    k_f32_to_bf16<<<blocks_for(ATT * HID), threads, 0, stream>>>(W_dst, Wdst_bf, ATT * HID);
    k_f32_to_bf16<<<blocks_for(ATT * HID), threads, 0, stream>>>(W_qr,  Wqr_bf,  ATT * HID);
    k_f32_to_bf16<<<blocks_for(HID * HID), threads, 0, stream>>>(W_msg, Wmsg_bf, HID * HID);
    k_f32_to_bf16<<<blocks_for(HID * HID), threads, 0, stream>>>(W_out, Wout_bf, HID * HID);

    // 3) line-graph attention + message + scatter (persistent waves,
    //    weights staged in LDS; 80 KB/block of 320 KB/WGP)
    {
        const int tiles  = (n_line + 15) / 16;
        int blocks = (tiles + 7) / 8;
        if (blocks > 2048) blocks = 2048;
        k_line<<<blocks, 256, 0, stream>>>(eh_bf, qr_bf, Wsrc_bf, Wdst_bf, Wqr_bf,
                                           b_qr, w_alpha, b_alpha, Wmsg_bf,
                                           line_src, line_dst, agg, n_line);
    }
    // 4) final GEMM: out = agg @ W_out^T (W_out staged in LDS)
    {
        const int tiles  = (n_edge + 15) / 16;
        int blocks = (tiles + 7) / 8;
        if (blocks > 512) blocks = 512;
        k_out<<<blocks, 256, 0, stream>>>(agg, Wout_bf, out, n_edge);
    }
}
